// ClofNet_vel_49357764165833
// MI455X (gfx1250) — matrix operand order repaired and checked by reading, not compiled, not run on hardware
//
#include <hip/hip_runtime.h>
#include <math.h>

#define NN 8192        // total nodes
#define NE 262144      // total edges
#define NGRAPH 128
#define NNODE 64
#define HID 128

typedef _Float16 h16;
typedef __attribute__((ext_vector_type(16))) _Float16 v16h;
typedef __attribute__((ext_vector_type(8)))  _Float16 v8h;
typedef __attribute__((ext_vector_type(8)))  float    v8f;

enum { MODE_F16 = 0, MODE_F32 = 1, MODE_EDGE = 2, MODE_NODE = 3 };

struct GemmArgs {
  const float* A32;     // MODE_F32 dense A (lda == K, K % 32 == 0)
  const h16*   A16;     // MODE_F16 dense A (lda == K, K % 32 == 0)
  int lda;
  const float* hbuf;    // MODE_EDGE / MODE_NODE: node features h [NN,HID] f32
  const float* radial;  // MODE_EDGE: [NE]
  const h16*   ef;      // MODE_EDGE: [NE,64] f16
  const int*   row;     // MODE_EDGE
  const int*   col;     // MODE_EDGE
  const float* agg;     // MODE_NODE: [NN,HID] f32
  const h16*   Wt;      // weights, transposed+padded: [Np, Kp] f16
  const float* bias;    // [N] f32 or nullptr
  float* C32;
  h16*   C16;
  int ldc;
  int M, N, Np, Kp;
};

__device__ __forceinline__ void pack8(v16h& a, int base, float4 u, float4 v) {
  a[base+0] = (h16)u.x; a[base+1] = (h16)u.y; a[base+2] = (h16)u.z; a[base+3] = (h16)u.w;
  a[base+4] = (h16)v.x; a[base+5] = (h16)v.y; a[base+6] = (h16)v.z; a[base+7] = (h16)v.w;
}

__device__ __forceinline__ v16h loadA_f32(const float* p) {
  float4 x0 = *(const float4*)(p);
  float4 x1 = *(const float4*)(p + 4);
  float4 x2 = *(const float4*)(p + 8);
  float4 x3 = *(const float4*)(p + 12);
  v16h a;
  pack8(a, 0, x0, x1);
  pack8(a, 8, x2, x3);
  return a;
}

__device__ __forceinline__ v16h loadA_f16(const h16* p) {
  v8h lo  = *(const v8h*)(p);
  v8h hi8 = *(const v8h*)(p + 8);
  v16h a;
  #pragma unroll
  for (int j = 0; j < 8; ++j) { a[j] = lo[j]; a[j + 8] = hi8[j]; }
  return a;
}

// Build one lane's A fragment (contiguous k = 16*hi + s order, 16x32 MxK tile).
// All paths are contiguous vector loads; no per-element guards.
// EDGE concat K order (weights permuted to match):
//   [h_row(0..127) | h_col(128..255) | ef(256..319) | radial(320) | pad(..351)]
template<int MODE>
__device__ __forceinline__ v16h buildA(int k0, int hi,
                                       const float* hr, const float* hc,
                                       const h16* a16p, float rad, const h16* efp) {
  const int kb = k0 + (hi ? 16 : 0);
  if (MODE == MODE_EDGE) {
    if (k0 < 2*HID) {
      const float* p = (k0 < HID) ? (hr + kb) : (hc + (kb - HID));
      return loadA_f32(p);
    } else if (k0 < 320) {
      return loadA_f16(efp + (kb - 2*HID));
    } else {
      v16h a;
      #pragma unroll
      for (int j = 0; j < 16; ++j) a[j] = (h16)0.f;
      a[0] = hi ? (h16)0.f : (h16)rad;   // K=320 holds radial (lanes 0-15 only)
      return a;
    }
  } else if (MODE == MODE_NODE) {
    const float* p = (k0 < HID) ? (hr + kb) : (hc + (kb - HID));
    return loadA_f32(p);
  } else if (MODE == MODE_F32) {
    return loadA_f32(hr + kb);
  } else { // MODE_F16
    return loadA_f16(a16p + kb);
  }
}

// Wave-level GEMM: each wave computes a 16 x (NT*16) output tile.
// NT is compile-time so the WMMA chain is straight-line (EXEC all-ones).
template<int MODE, int NT, bool RELU, bool OUT16, bool ADDC>
__global__ void gemm_wmma(GemmArgs g) {
  const int wave = blockIdx.x * (blockDim.x >> 5) + (threadIdx.x >> 5);
  const int lane = threadIdx.x & 31;
  const int nSuper = g.Np / (NT * 16);
  const int mTiles = g.M >> 4;
  if (wave >= mTiles * nSuper) return;          // wave-uniform
  const int mt = wave / nSuper;
  const int sc = wave % nSuper;
  const int m0 = mt << 4;
  const int n0base = sc * (NT * 16);

  const int lm = lane & 15;
  const int hi = lane >> 4;
  const int am = m0 + lm;                       // A row for this lane

  // hoisted per-lane A row context
  const float* hr = nullptr; const float* hc = nullptr;
  const h16* a16p = nullptr; const h16* efp = nullptr; float rad = 0.f;
  if (MODE == MODE_EDGE) {
    hr = g.hbuf + (size_t)g.row[am] * HID;
    hc = g.hbuf + (size_t)g.col[am] * HID;
    rad = g.radial[am];
    efp = g.ef + (size_t)am * 64;
  } else if (MODE == MODE_NODE) {
    hr = g.hbuf + (size_t)am * HID;
    hc = g.agg + (size_t)am * HID;
  } else if (MODE == MODE_F32) {
    hr = g.A32 + (size_t)am * g.lda;
  } else {
    a16p = g.A16 + (size_t)am * g.lda;
  }

  v8f acc[NT];
  #pragma unroll
  for (int t = 0; t < NT; ++t) {
    float bv = 0.f;
    const int n = n0base + t*16 + lm;
    if (g.bias && (NT != 1 || n < g.N)) bv = g.bias[n];
    #pragma unroll
    for (int r = 0; r < 8; ++r) acc[t][r] = bv;
  }

  const h16* wBase = g.Wt + (size_t)(n0base + lm) * g.Kp;
  for (int k0 = 0; k0 < g.Kp; k0 += 32) {
    const v16h a = buildA<MODE>(k0, hi, hr, hc, a16p, rad, efp);
    const int kbB = k0 + (hi << 4);
    v16h bf[NT];
    #pragma unroll
    for (int t = 0; t < NT; ++t)
      bf[t] = *(const v16h*)(wBase + (size_t)t * 16 * g.Kp + kbB);
    #pragma unroll
    for (int t = 0; t < NT; ++t)
      acc[t] = __builtin_amdgcn_wmma_f32_16x16x32_f16(
          false, a, false, bf[t], (short)0, acc[t], false, false);
  }

  #pragma unroll
  for (int t = 0; t < NT; ++t) {
    const int n = n0base + t*16 + lm;
    if (NT == 1 && n >= g.N) continue;          // only pads when Np > N
    #pragma unroll
    for (int r = 0; r < 8; ++r) {
      const int m = m0 + r + (hi << 3);
      float v = acc[t][r];
      if (RELU) v = v > 0.f ? v : 0.f;
      if (OUT16) {
        g.C16[(size_t)m * g.ldc + n] = (h16)v;
      } else if (ADDC) {
        g.C32[(size_t)m * g.ldc + n] += v;
      } else {
        g.C32[(size_t)m * g.ldc + n] = v;
      }
    }
  }
}

// ---- small support kernels -------------------------------------------------

__global__ void k_prep_wt(const float* __restrict__ src, h16* __restrict__ dst,
                          int K, int N, int Kp, int Np) {
  int idx = blockIdx.x * blockDim.x + threadIdx.x;
  if (idx >= Np * Kp) return;
  int n = idx / Kp, k = idx % Kp;
  float v = (n < N && k < K) ? src[(size_t)k * N + n] : 0.f;  // transpose [K,N]->[Np,Kp]
  dst[idx] = (h16)v;
}

// e1 weights [321,128] -> [128,352] with K remap matching the EDGE concat
// permutation: k'<256 -> k', 256..319 -> ef rows 257..320, 320 -> radial row 256.
__global__ void k_prep_wt_e1(const float* __restrict__ src, h16* __restrict__ dst) {
  int idx = blockIdx.x * blockDim.x + threadIdx.x;
  if (idx >= 128 * 352) return;
  int n = idx / 352, k = idx % 352;
  int ks = (k < 256) ? k : (k < 320) ? (257 + (k - 256)) : (k == 320) ? 256 : -1;
  float v = (ks >= 0) ? src[(size_t)ks * 128 + n] : 0.f;
  dst[idx] = (h16)v;
}

// h_in [NN,16] f32 -> zero-padded [NN,32] f16
__global__ void k_h16pad(const float* __restrict__ src, h16* __restrict__ dst) {
  int i = blockIdx.x * blockDim.x + threadIdx.x;
  if (i >= NN * 32) return;
  int m = i >> 5, k = i & 31;
  dst[i] = (h16)((k < 16) ? src[m * 16 + k] : 0.f);
}

__global__ void k_zero(float* p, int n) {
  int i = blockIdx.x * blockDim.x + threadIdx.x;
  if (i < n) p[i] = 0.f;
}

__global__ void k_deg(const int* __restrict__ row, float* deg) {
  int e = blockIdx.x * blockDim.x + threadIdx.x;
  if (e < NE) atomicAdd(&deg[row[e]], 1.f);
}

__global__ void k_centroid(const float* __restrict__ x, float* xc, float* cent) {
  __shared__ float s0[64], s1[64], s2[64];
  int g = blockIdx.x, t = threadIdx.x;
  int n = g * NNODE + t;
  float a0 = x[n*3+0], a1 = x[n*3+1], a2 = x[n*3+2];
  s0[t] = a0; s1[t] = a1; s2[t] = a2;
  __syncthreads();
  for (int st = 32; st > 0; st >>= 1) {
    if (t < st) { s0[t] += s0[t+st]; s1[t] += s1[t+st]; s2[t] += s2[t+st]; }
    __syncthreads();
  }
  float c0 = s0[0] * (1.f/NNODE), c1 = s1[0] * (1.f/NNODE), c2 = s2[0] * (1.f/NNODE);
  if (t == 0) { cent[g*3+0] = c0; cent[g*3+1] = c1; cent[g*3+2] = c2; }
  xc[n*3+0] = a0 - c0; xc[n*3+1] = a1 - c1; xc[n*3+2] = a2 - c2;
}

__device__ __forceinline__ void edge_frame(const float* xc, int r, int c,
    float& radial, float& d0, float& d1, float& d2,
    float& q0, float& q1, float& q2, float& v0, float& v1, float& v2) {
  float xr0 = xc[r*3+0], xr1 = xc[r*3+1], xr2 = xc[r*3+2];
  float xq0 = xc[c*3+0], xq1 = xc[c*3+1], xq2 = xc[c*3+2];
  d0 = xr0 - xq0; d1 = xr1 - xq1; d2 = xr2 - xq2;
  radial = d0*d0 + d1*d1 + d2*d2;
  q0 = xr1*xq2 - xr2*xq1;                    // cross(xr, xq)
  q1 = xr2*xq0 - xr0*xq2;
  q2 = xr0*xq1 - xr1*xq0;
  float invd = 1.f / (sqrtf(radial) + 1.f);
  d0 *= invd; d1 *= invd; d2 *= invd;
  float invq = 1.f / (sqrtf(q0*q0 + q1*q1 + q2*q2) + 1.f);
  q0 *= invq; q1 *= invq; q2 *= invq;
  v0 = d1*q2 - d2*q1;                        // vert = cross(d, cross)
  v1 = d2*q0 - d0*q2;
  v2 = d0*q1 - d1*q0;
}

// writes zero-padded [NE,32] f16 fuse-input rows:
// [ea(4), sin, cos, coff_i(3), coff_j(3), vel_i(3), vel_j(3), 0 x 14]
__global__ void k_edge_feat(const float* __restrict__ xc, const float* __restrict__ vel,
                            const float* __restrict__ ea,
                            const int* __restrict__ row, const int* __restrict__ col,
                            h16* __restrict__ feat) {
  int e = blockIdx.x * blockDim.x + threadIdx.x;
  if (e >= NE) return;
  int r = row[e], c = col[e];
  float radial, d0,d1,d2, q0,q1,q2, v0,v1,v2;
  edge_frame(xc, r, c, radial, d0,d1,d2, q0,q1,q2, v0,v1,v2);
  float xr0 = xc[r*3+0], xr1 = xc[r*3+1], xr2 = xc[r*3+2];
  float xq0 = xc[c*3+0], xq1 = xc[c*3+1], xq2 = xc[c*3+2];
  float vr0 = vel[r*3+0], vr1 = vel[r*3+1], vr2 = vel[r*3+2];
  float vq0 = vel[c*3+0], vq1 = vel[c*3+1], vq2 = vel[c*3+2];
  float ci0 = d0*xr0 + d1*xr1 + d2*xr2;
  float ci1 = q0*xr0 + q1*xr1 + q2*xr2;
  float ci2 = v0*xr0 + v1*xr1 + v2*xr2;
  float cj0 = d0*xq0 + d1*xq1 + d2*xq2;
  float cj1 = q0*xq0 + q1*xq1 + q2*xq2;
  float cj2 = v0*xq0 + v1*xq1 + v2*xq2;
  float wi0 = d0*vr0 + d1*vr1 + d2*vr2;
  float wi1 = q0*vr0 + q1*vr1 + q2*vr2;
  float wi2 = v0*vr0 + v1*vr1 + v2*vr2;
  float wj0 = d0*vq0 + d1*vq1 + d2*vq2;
  float wj1 = q0*vq0 + q1*vq1 + q2*vq2;
  float wj2 = v0*vq0 + v1*vq1 + v2*vq2;
  float ni = sqrtf(ci0*ci0 + ci1*ci1 + ci2*ci2);
  float nj = sqrtf(cj0*cj0 + cj1*cj1 + cj2*cj2);
  float cosv = (ci0*cj0 + ci1*cj1 + ci2*cj2) / (ni + 1e-5f) / (nj + 1e-5f);
  float s2 = 1.f - cosv*cosv; if (s2 < 0.f) s2 = 0.f;
  float sinv = sqrtf(s2);
  h16* f = feat + (size_t)e * 32;
  f[0] = (h16)ea[e*4+0]; f[1] = (h16)ea[e*4+1]; f[2] = (h16)ea[e*4+2]; f[3] = (h16)ea[e*4+3];
  f[4] = (h16)sinv; f[5] = (h16)cosv;
  f[6] = (h16)ci0; f[7] = (h16)ci1; f[8] = (h16)ci2;
  f[9] = (h16)cj0; f[10] = (h16)cj1; f[11] = (h16)cj2;
  f[12] = (h16)wi0; f[13] = (h16)wi1; f[14] = (h16)wi2;
  f[15] = (h16)wj0; f[16] = (h16)wj1; f[17] = (h16)wj2;
  #pragma unroll
  for (int k = 18; k < 32; ++k) f[k] = (h16)0.f;
}

__global__ void k_frame(const float* __restrict__ xc,
                        const int* __restrict__ row, const int* __restrict__ col,
                        float* __restrict__ radialB, float* __restrict__ basis) {
  int e = blockIdx.x * blockDim.x + threadIdx.x;
  if (e >= NE) return;
  float radial, d0,d1,d2, q0,q1,q2, v0,v1,v2;
  edge_frame(xc, row[e], col[e], radial, d0,d1,d2, q0,q1,q2, v0,v1,v2);
  radialB[e] = radial;
  float* b = basis + (size_t)e * 9;
  b[0]=d0; b[1]=d1; b[2]=d2; b[3]=q0; b[4]=q1; b[5]=q2; b[6]=v0; b[7]=v1; b[8]=v2;
}

__global__ void k_trans(const float* __restrict__ basis, const float* __restrict__ coff,
                        const int* __restrict__ row, float* transAcc) {
  int e = blockIdx.x * blockDim.x + threadIdx.x;
  if (e >= NE) return;
  const float* b = basis + (size_t)e * 9;
  float c0 = coff[e*3+0], c1 = coff[e*3+1], c2 = coff[e*3+2];
  int r = row[e];
  #pragma unroll
  for (int k = 0; k < 3; ++k) {
    float t = b[k]*c0 + b[3+k]*c1 + b[6+k]*c2;
    t = fminf(100.f, fmaxf(-100.f, t));
    atomicAdd(&transAcc[r*3+k], t);
  }
}

__global__ void k_scatter_m(const h16* __restrict__ m, const int* __restrict__ row,
                            float* agg) {
  size_t i = (size_t)blockIdx.x * blockDim.x + threadIdx.x;
  if (i >= (size_t)NE * HID) return;
  int e = (int)(i >> 7);
  int f = (int)(i & 127);
  atomicAdd(&agg[(size_t)row[e] * HID + f], (float)m[i]);
}

__global__ void k_update_x(float* xc, const float* __restrict__ transAcc,
                           const float* __restrict__ deg,
                           const float* __restrict__ vscale,
                           const float* __restrict__ vel) {
  int i = blockIdx.x * blockDim.x + threadIdx.x;
  if (i >= NN) return;
  float c = deg[i]; if (c < 1.f) c = 1.f;
  float inv = 1.f / c;
  float vs = vscale[i];
  #pragma unroll
  for (int d = 0; d < 3; ++d)
    xc[i*3+d] += transAcc[i*3+d] * inv + vs * vel[i*3+d];
}

__global__ void k_xout(const float* __restrict__ xc, const float* __restrict__ cent,
                       float* xout) {
  int i = blockIdx.x * blockDim.x + threadIdx.x;
  if (i >= NN) return;
  int g = i / NNODE;
  #pragma unroll
  for (int d = 0; d < 3; ++d) xout[i*3+d] = xc[i*3+d] + cent[g*3+d];
}

__global__ void k_dist(const float* __restrict__ xo, float* dist) {
  int j = blockIdx.x * 16 + (threadIdx.x & 15);
  int i = blockIdx.y * 16 + (threadIdx.x >> 4);
  float xi0 = xo[i*3+0], xi1 = xo[i*3+1], xi2 = xo[i*3+2];
  float xj0 = xo[j*3+0], xj1 = xo[j*3+1], xj2 = xo[j*3+2];
  float si = xi0*xi0 + xi1*xi1 + xi2*xi2;
  float sj = xj0*xj0 + xj1*xj1 + xj2*xj2;
  float dd = si + sj - 2.f * (xi0*xj0 + xi1*xj1 + xi2*xj2);
  if (dd < 0.f) dd = 0.f;
  dist[(size_t)i * NN + j] = (dd > 1e-12f) ? sqrtf(dd) : 0.f;
}

// ---- host ------------------------------------------------------------------

template<int MODE, int NT, bool RELU, bool OUT16, bool ADDC>
static void launch_gemm(const GemmArgs& g, hipStream_t s) {
  int waves = (g.M >> 4) * (g.Np / (NT * 16));
  int blocks = (waves + 7) / 8;
  gemm_wmma<MODE, NT, RELU, OUT16, ADDC><<<blocks, 256, 0, s>>>(g);
}

extern "C" void kernel_launch(void* const* d_in, const int* in_sizes, int n_in,
                              void* d_out, int out_size, void* d_ws, size_t ws_size,
                              hipStream_t stream) {
  (void)in_sizes; (void)n_in; (void)out_size; (void)d_ws; (void)ws_size;

  const float* h_in   = (const float*)d_in[0];   // [NN,16]
  const float* x_in   = (const float*)d_in[1];   // [NN,3]
  const float* vel_in = (const float*)d_in[2];   // [NN,3]
  const float* ea_in  = (const float*)d_in[3];   // [NE,4]
  const int*   edges  = (const int*)d_in[4];     // [2,NE]
  const int* row = edges;
  const int* col = edges + NE;
  // d_in[5] = n_nodes scalar (64), hardcoded.

  // params flattened in jax tree order (dict keys sorted):
  // emb(w,b), fuse1(w,b), fuse2(w,b), layers[l]: c1(w,b), c2(w), e1(w,b),
  // e2(w,b), n1(w,b), n2(w,b), v1(w,b), v2(w,b)
  int pi = 6;
  const float* emb_w = (const float*)d_in[pi++]; const float* emb_b = (const float*)d_in[pi++];
  const float* f1_w  = (const float*)d_in[pi++]; const float* f1_b  = (const float*)d_in[pi++];
  const float* f2_w  = (const float*)d_in[pi++]; const float* f2_b  = (const float*)d_in[pi++];
  struct LayerP { const float *c1w,*c1b,*c2w,*e1w,*e1b,*e2w,*e2b,*n1w,*n1b,*n2w,*n2b,*v1w,*v1b,*v2w,*v2b; } L[4];
  for (int l = 0; l < 4; ++l) {
    L[l].c1w = (const float*)d_in[pi++]; L[l].c1b = (const float*)d_in[pi++];
    L[l].c2w = (const float*)d_in[pi++];
    L[l].e1w = (const float*)d_in[pi++]; L[l].e1b = (const float*)d_in[pi++];
    L[l].e2w = (const float*)d_in[pi++]; L[l].e2b = (const float*)d_in[pi++];
    L[l].n1w = (const float*)d_in[pi++]; L[l].n1b = (const float*)d_in[pi++];
    L[l].n2w = (const float*)d_in[pi++]; L[l].n2b = (const float*)d_in[pi++];
    L[l].v1w = (const float*)d_in[pi++]; L[l].v1b = (const float*)d_in[pi++];
    L[l].v2w = (const float*)d_in[pi++]; L[l].v2b = (const float*)d_in[pi++];
  }

  float* out   = (float*)d_out;
  float* hA    = out;                         // [NN,HID], working h lives here
  float* xoutA = out + (size_t)NN * HID;      // [NN,3]
  float* distA = xoutA + (size_t)NN * 3;      // [NN,NN] — scratch until the end

  // All large scratch lives inside the dist region (67.1M floats, ~52M used),
  // fully overwritten by k_dist at the end.
  size_t soff = 0;
  auto takeS = [&](size_t elemsFloat) -> float* {
    float* p = distA + soff;
    soff += (elemsFloat + 63) & ~(size_t)63;
    return p;
  };
  h16*   feat32   = (h16*)takeS((size_t)NE * 16);    // [NE,32] f16 (padded)
  h16*   efbuf    = (h16*)takeS((size_t)NE * 32);    // [NE,64] f16
  float* radialB  = takeS((size_t)NE);
  float* basisB   = takeS((size_t)NE * 9);
  h16*   ehid     = (h16*)takeS((size_t)NE * 64);    // [NE,128] f16 (also fuse hidden)
  h16*   mbuf     = (h16*)takeS((size_t)NE * 64);    // [NE,128] f16
  float* coffB    = takeS((size_t)NE * 3);
  float* aggB     = takeS((size_t)NN * HID);
  float* xcB      = takeS((size_t)NN * 3);
  float* centB    = takeS((size_t)NGRAPH * 3);
  float* degB     = takeS((size_t)NN);
  float* transAcc = takeS((size_t)NN * 3);
  h16*   nhid     = (h16*)takeS((size_t)NN * 64);    // [NN,128] f16
  float* vscale   = takeS((size_t)NN);
  h16*   h16pad   = (h16*)takeS((size_t)NN * 16);    // [NN,32] f16 (padded h_in)

  auto prep = [&](const float* src, int K, int N, int Kp, int Np) -> h16* {
    h16* dst = (h16*)takeS((size_t)Np * Kp / 2);
    int tot = Np * Kp;
    k_prep_wt<<<(tot + 255) / 256, 256, 0, stream>>>(src, dst, K, N, Kp, Np);
    return dst;
  };

  h16* wt_emb = prep(emb_w, 16, 128, 32, 128);
  h16* wt_f1  = prep(f1_w, 18, 64, 32, 64);
  h16* wt_f2  = prep(f2_w, 64, 64, 64, 64);
  h16 *wt_e1[4], *wt_e2[4], *wt_n1[4], *wt_n2[4], *wt_c1[4], *wt_c2[4], *wt_v1[4], *wt_v2[4];
  for (int l = 0; l < 4; ++l) {
    wt_e1[l] = (h16*)takeS((size_t)128 * 352 / 2);
    k_prep_wt_e1<<<(128 * 352 + 255) / 256, 256, 0, stream>>>(L[l].e1w, wt_e1[l]);
    wt_e2[l] = prep(L[l].e2w, 128, 128, 128, 128);
    wt_n1[l] = prep(L[l].n1w, 256, 128, 256, 128);
    wt_n2[l] = prep(L[l].n2w, 128, 128, 128, 128);
    wt_c1[l] = prep(L[l].c1w, 128, 128, 128, 128);
    wt_c2[l] = prep(L[l].c2w, 128, 3,   128, 16);
    wt_v1[l] = prep(L[l].v1w, 128, 128, 128, 128);
    wt_v2[l] = prep(L[l].v2w, 128, 1,   128, 16);
  }

  // degree (static across layers) + centroid/centered coords + padded h_in
  k_zero<<<(NN + 255) / 256, 256, 0, stream>>>(degB, NN);
  k_deg<<<(NE + 255) / 256, 256, 0, stream>>>(row, degB);
  k_centroid<<<NGRAPH, NNODE, 0, stream>>>(x_in, xcB, centB);
  k_h16pad<<<(NN * 32 + 255) / 256, 256, 0, stream>>>(h_in, h16pad);

  // h = h_in @ emb_w + emb_b  -> hA (f32)
  { GemmArgs g{}; g.A16 = h16pad; g.lda = 32;
    g.Wt = wt_emb; g.bias = emb_b; g.C32 = hA; g.ldc = HID;
    g.M = NN; g.N = HID; g.Np = 128; g.Kp = 32;
    launch_gemm<MODE_F16, 4, false, false, false>(g, stream); }

  // scalarization + fuse MLP -> ef [NE,64] f16
  k_edge_feat<<<(NE + 255) / 256, 256, 0, stream>>>(xcB, vel_in, ea_in, row, col, feat32);
  { GemmArgs g{}; g.A16 = feat32; g.lda = 32;
    g.Wt = wt_f1; g.bias = f1_b; g.C16 = ehid; g.ldc = 64;
    g.M = NE; g.N = 64; g.Np = 64; g.Kp = 32;
    launch_gemm<MODE_F16, 4, true, true, false>(g, stream); }
  { GemmArgs g{}; g.A16 = ehid; g.lda = 64;
    g.Wt = wt_f2; g.bias = f2_b; g.C16 = efbuf; g.ldc = 64;
    g.M = NE; g.N = 64; g.Np = 64; g.Kp = 64;
    launch_gemm<MODE_F16, 4, true, true, false>(g, stream); }

  for (int l = 0; l < 4; ++l) {
    k_frame<<<(NE + 255) / 256, 256, 0, stream>>>(xcB, row, col, radialB, basisB);

    // m = relu(relu([h_r,h_c,ef,radial]@e1+b)@e2+b)   (K-permuted concat)
    { GemmArgs g{}; g.hbuf = hA; g.radial = radialB; g.ef = efbuf;
      g.row = row; g.col = col;
      g.Wt = wt_e1[l]; g.bias = L[l].e1b; g.C16 = ehid; g.ldc = HID;
      g.M = NE; g.N = HID; g.Np = 128; g.Kp = 352;
      launch_gemm<MODE_EDGE, 4, true, true, false>(g, stream); }
    { GemmArgs g{}; g.A16 = ehid; g.lda = HID;
      g.Wt = wt_e2[l]; g.bias = L[l].e2b; g.C16 = mbuf; g.ldc = HID;
      g.M = NE; g.N = HID; g.Np = 128; g.Kp = 128;
      launch_gemm<MODE_F16, 4, true, true, false>(g, stream); }

    // coff = relu(m@c1+b)@c2
    { GemmArgs g{}; g.A16 = mbuf; g.lda = HID;
      g.Wt = wt_c1[l]; g.bias = L[l].c1b; g.C16 = ehid; g.ldc = HID;
      g.M = NE; g.N = HID; g.Np = 128; g.Kp = 128;
      launch_gemm<MODE_F16, 4, true, true, false>(g, stream); }
    { GemmArgs g{}; g.A16 = ehid; g.lda = HID;
      g.Wt = wt_c2[l]; g.bias = nullptr; g.C32 = coffB; g.ldc = 3;
      g.M = NE; g.N = 3; g.Np = 16; g.Kp = 128;
      launch_gemm<MODE_F16, 1, false, false, false>(g, stream); }

    // segment sums
    k_zero<<<(NN * 3 + 255) / 256, 256, 0, stream>>>(transAcc, NN * 3);
    k_zero<<<(NN * HID + 255) / 256, 256, 0, stream>>>(aggB, NN * HID);
    k_trans<<<(NE + 255) / 256, 256, 0, stream>>>(basisB, coffB, row, transAcc);
    {
      size_t tot = (size_t)NE * HID;
      k_scatter_m<<<(unsigned)((tot + 255) / 256), 256, 0, stream>>>(mbuf, row, aggB);
    }

    // vscale = relu(h@v1+b)@v2+b  (uses pre-update h)
    { GemmArgs g{}; g.A32 = hA; g.lda = HID;
      g.Wt = wt_v1[l]; g.bias = L[l].v1b; g.C16 = nhid; g.ldc = HID;
      g.M = NN; g.N = HID; g.Np = 128; g.Kp = 128;
      launch_gemm<MODE_F32, 4, true, true, false>(g, stream); }
    { GemmArgs g{}; g.A16 = nhid; g.lda = HID;
      g.Wt = wt_v2[l]; g.bias = L[l].v2b; g.C32 = vscale; g.ldc = 1;
      g.M = NN; g.N = 1; g.Np = 16; g.Kp = 128;
      launch_gemm<MODE_F16, 1, false, false, false>(g, stream); }

    k_update_x<<<(NN + 255) / 256, 256, 0, stream>>>(xcB, transAcc, degB, vscale, vel_in);

    // h += relu([h,agg]@n1+b)@n2+b
    { GemmArgs g{}; g.hbuf = hA; g.agg = aggB;
      g.Wt = wt_n1[l]; g.bias = L[l].n1b; g.C16 = nhid; g.ldc = HID;
      g.M = NN; g.N = HID; g.Np = 128; g.Kp = 256;
      launch_gemm<MODE_NODE, 4, true, true, false>(g, stream); }
    { GemmArgs g{}; g.A16 = nhid; g.lda = HID;
      g.Wt = wt_n2[l]; g.bias = L[l].n2b; g.C32 = hA; g.ldc = HID;
      g.M = NN; g.N = HID; g.Np = 128; g.Kp = 128;
      launch_gemm<MODE_F16, 4, false, false, true>(g, stream); }
  }

  // outputs: xout then dist (dist overwrite frees all scratch above)
  k_xout<<<(NN + 255) / 256, 256, 0, stream>>>(xcB, centB, xoutA);
  k_dist<<<dim3(NN / 16, NN / 16), 256, 0, stream>>>(xoutA, distA);
}
